// TopologicalDecoder_27195732918749
// MI455X (gfx1250) — matrix operand order repaired and checked by reading, not compiled, not run on hardware
//
#include <hip/hip_runtime.h>
#include <math.h>

// Problem constants (reference: B=32768, L=64, H=512, C=8, O=512)
#define LDIM 64
#define HDIM 512
#define CDIM 8
#define ODIM 512
#define ROWS 64            // batch rows per block (4 M-tiles of 16)
#define NTPW 2             // n-tiles per wave (16 waves * 2 * 16 cols = 512)
#define BLOCK_THREADS 512  // 16 waves (wave32)

typedef __attribute__((ext_vector_type(16))) __bf16    v16bf;
typedef __attribute__((ext_vector_type(8)))  float     v8f;
typedef __attribute__((ext_vector_type(4)))  unsigned  v4u;

union Frag {
    v16bf    v;
    unsigned u[8];
    v4u      q[2];
};

__device__ __forceinline__ unsigned short f2bf(float f) {
    union { float f; unsigned u; } x;
    x.f = f;
    unsigned r = x.u + 0x7FFFu + ((x.u >> 16) & 1u);  // round-to-nearest-even
    return (unsigned short)(r >> 16);
}

__device__ __forceinline__ unsigned pack2(float a, float b) {
    return (unsigned)f2bf(a) | ((unsigned)f2bf(b) << 16);
}

__device__ __forceinline__ float gelu_exact(float x) {
    return 0.5f * x * (1.0f + erff(x * 0.70710678118654752440f));
}

// Load a (k, k+1) bf16 pair: fp32 source converts, bf16 source is one dword.
__device__ __forceinline__ unsigned load_pair(const float* __restrict__ p) {
    return pack2(p[0], p[1]);
}
__device__ __forceinline__ unsigned load_pair(const unsigned short* __restrict__ p) {
    return *reinterpret_cast<const unsigned*>(p);
}

// ---- Swizzled (fragment-major) LDS layout for A operands --------------------
// Fragment for (mt, ks, lane) is 32 contiguous bytes; index of element (m,k)
// in bf16 units. Contiguous-k A layout (k = 16h + s, same order as the B loaders):
//   lane = (m&15) + 16*h with h=kk>>4; dword v = (kk&15)>>1.
__device__ __forceinline__ int swz16(int m, int k, int ksteps) {
    const int mt = m >> 4, mm = m & 15;
    const int ks = k >> 5, kk = k & 31;
    const int h  = kk >> 4;
    const int v  = (kk & 15) >> 1;
    const int ln = mm + (h << 4);
    return ((((mt * ksteps + ks) * 32 + ln) * 8 + v) << 1) + (kk & 1);
}

// A-fragment: two ds_load_b128 from the swizzled buffer.
__device__ __forceinline__ Frag load_a_swz(const unsigned short* A, int mt,
                                           int ksteps, int ks, int lane) {
    const unsigned* p = reinterpret_cast<const unsigned*>(A) +
                        ((size_t)((mt * ksteps + ks) * 32 + lane) * 8);
    Frag f;
    f.q[0] = *reinterpret_cast<const v4u*>(p);
    f.q[1] = *reinterpret_cast<const v4u*>(p + 4);
    return f;
}

// B-matrix (32x16) from global row-major weight W[n][k] (n-row stride ldk).
// lane n=lane&15, half=lane>>4; VGPR v -> K = 16*half + 2v, 2v+1 (contiguous 32B).
template <typename WT>
__device__ __forceinline__ Frag load_b_rm(const WT* __restrict__ W, int ldk,
                                          int col, int lane, int kbase) {
    const int half = lane >> 4;
    const WT* __restrict__ row = W + (size_t)col * ldk + kbase + 16 * half;
    Frag f;
#pragma unroll
    for (int v = 0; v < 8; ++v) f.u[v] = load_pair(row + 2 * v);
    return f;
}

// B-matrix for the chart GEMM: element (k, n) = chart_w[c][n][l], c=k>>6, l=k&63.
// The 16-element run starting at a multiple of 16 stays inside one chart.
template <typename WT>
__device__ __forceinline__ Frag load_b_chart(const WT* __restrict__ W,
                                             int col, int lane, int kbase) {
    const int half = lane >> 4;
    const int k0 = kbase + 16 * half;
    const int c = k0 >> 6, l = k0 & 63;
    const WT* __restrict__ p = W + ((size_t)(c * HDIM + col) * LDIM + l);
    Frag f;
#pragma unroll
    for (int v = 0; v < 8; ++v) f.u[v] = load_pair(p + 2 * v);
    return f;
}

__device__ __forceinline__ v8f wmma_bf16(const Frag& a, const Frag& b, v8f c) {
    return __builtin_amdgcn_wmma_f32_16x16x32_bf16(
        /*neg_a=*/false, a.v, /*neg_b=*/false, b.v,
        /*c_mod=*/(short)0, c, /*reuse_a=*/false, /*reuse_b=*/false);
}

// One GEMM chain: 4 M-tiles x 1 N-tile, B double-buffered one k-step ahead,
// all 4 A-fragments batched per k-step (one DS clause, 4 back-to-back WMMAs).
template <typename WT, bool CHART>
__device__ __forceinline__ void gemm_chain(v8f* acc, const unsigned short* Abase,
                                           int aksteps, const WT* __restrict__ W,
                                           int ldk, int nks, int col, int lane) {
    Frag b = CHART ? load_b_chart(W, col, lane, 0)
                   : load_b_rm(W, ldk, col, lane, 0);
#pragma unroll 1
    for (int ks = 0; ks < nks; ++ks) {
        Frag a0 = load_a_swz(Abase, 0, aksteps, ks, lane);
        Frag a1 = load_a_swz(Abase, 1, aksteps, ks, lane);
        Frag a2 = load_a_swz(Abase, 2, aksteps, ks, lane);
        Frag a3 = load_a_swz(Abase, 3, aksteps, ks, lane);
        if (!CHART && (ks + 2) < nks)   // gfx1250 global_prefetch_b8 on weights
            __builtin_prefetch((const void*)(W + (size_t)col * ldk + (ks + 2) * 32 +
                                             16 * (lane >> 4)), 0, 0);
        const int ksn = (ks + 1 < nks) ? (ks + 1) : ks;
        Frag bn = CHART ? load_b_chart(W, col, lane, ksn * 32)
                        : load_b_rm(W, ldk, col, lane, ksn * 32);
        acc[0] = wmma_bf16(a0, b, acc[0]);
        acc[1] = wmma_bf16(a1, b, acc[1]);
        acc[2] = wmma_bf16(a2, b, acc[2]);
        acc[3] = wmma_bf16(a3, b, acc[3]);
        b = bn;
    }
}

// ---- Weight pre-convert (fp32 -> packed bf16 in workspace) ------------------
__global__ void __launch_bounds__(256)
convert_w_kernel(const float* __restrict__ src, unsigned short* __restrict__ dst, int n) {
    const int i = blockIdx.x * 256 + threadIdx.x;
    if (i < n) dst[i] = f2bf(src[i]);
}

// ---- Fused decoder kernel ---------------------------------------------------
// One block = 512 threads = 16 waves, handles 64 batch rows (4 M-tiles).
// Each wave owns 2 N-tiles; B-fragments are reused across the 4 M-tiles.
template <typename WT>
__global__ void __launch_bounds__(BLOCK_THREADS)
topo_decoder_kernel(const float* __restrict__ z_geo, const float* __restrict__ z_tex,
                    const WT* __restrict__ chart_w, const float* __restrict__ chart_b,
                    const float* __restrict__ router_w, const float* __restrict__ router_b,
                    const WT* __restrict__ mlp1_w, const float* __restrict__ mlp1_b,
                    const WT* __restrict__ mlp2_w, const float* __restrict__ mlp2_b,
                    const WT* __restrict__ skip_w, const float* __restrict__ skip_b,
                    const WT* __restrict__ tex_w, const float* __restrict__ tex_b,
                    const float* __restrict__ tex_scale_p,
                    float* __restrict__ out_x, float* __restrict__ out_r) {
    // Dynamic LDS partition (total 225280 B; gfx1250 allows 320KB/workgroup)
    extern __shared__ char smem[];
    unsigned short* bufA  = (unsigned short*)(smem);            // 64KB: A' (ph1) / h1 (ph3), swizzled
    unsigned short* h_bf  = (unsigned short*)(smem + 65536);    // 64KB: h_global, swizzled
    unsigned short* g1_bf = (unsigned short*)(smem + 131072);   // 64KB: gelu(h_global), swizzled
    float*          z_f   = (float*)(smem + 196608);            // 16KB: tanh(z_geo)
    unsigned short* zt_bf = (unsigned short*)(smem + 212992);   //  8KB: tanh(z_tex), swizzled
    float*          r_f   = (float*)(smem + 221184);            //  2KB: router weights
    float*          lg_s  = (float*)(smem + 223232);            //  2KB: logits

    const int tid  = threadIdx.x;
    const int lane = tid & 31;
    const int wv   = tid >> 5;          // 0..15
    const int half = lane >> 4;
    const int nn   = lane & 15;
    const int row0 = blockIdx.x * ROWS;
    const float tscale = *tex_scale_p;

    // Phase 0: load + tanh activations
    for (int i = tid; i < ROWS * LDIM; i += BLOCK_THREADS) {
        const int m = i >> 6, l = i & 63;
        z_f[i] = tanhf(z_geo[(size_t)(row0 + m) * LDIM + l]);
    }
    for (int i = tid; i < ROWS * LDIM; i += BLOCK_THREADS) {
        const int m = i >> 6, l = i & 63;
        zt_bf[swz16(m, l, 2)] = f2bf(tanhf(z_tex[(size_t)(row0 + m) * LDIM + l]));
    }
    __syncthreads();

    // Router logits: 64 rows x 8 charts = 512, one per thread
    {
        const int m = tid >> 3, c = tid & 7;
        float s = router_b[c];
        const float* rw = router_w + c * LDIM;
#pragma unroll 8
        for (int l = 0; l < LDIM; ++l) s += z_f[m * LDIM + l] * rw[l];
        lg_s[m * CDIM + c] = s;
    }
    __syncthreads();

    // Softmax over C=8, one row per thread; emit router output
    if (tid < ROWS) {
        const int m = tid;
        float mx = lg_s[m * CDIM];
#pragma unroll
        for (int c = 1; c < CDIM; ++c) mx = fmaxf(mx, lg_s[m * CDIM + c]);
        float e[CDIM], sum = 0.0f;
#pragma unroll
        for (int c = 0; c < CDIM; ++c) { e[c] = expf(lg_s[m * CDIM + c] - mx); sum += e[c]; }
        const float inv = 1.0f / sum;
#pragma unroll
        for (int c = 0; c < CDIM; ++c) {
            const float rv = e[c] * inv;
            r_f[m * CDIM + c] = rv;
            out_r[(size_t)(row0 + m) * CDIM + c] = rv;
        }
    }
    __syncthreads();

    // Build A'[m][c*64+l] = r[m][c] * z[m][l] (folds chart mixture into one GEMM)
    for (int i = tid; i < ROWS * 512; i += BLOCK_THREADS) {
        const int m = i >> 9, k = i & 511;
        bufA[swz16(m, k, 16)] = f2bf(r_f[m * CDIM + (k >> 6)] * z_f[m * LDIM + (k & 63)]);
    }
    __syncthreads();

    // ---- Phase 1: h_global = A' @ reshape(chart_w), [64x512] x [512x512] ----
#pragma unroll 1
    for (int tt = 0; tt < NTPW; ++tt) {
        const int col = ((wv * NTPW + tt) << 4) + nn;
        v8f acc[4];
#pragma unroll
        for (int i = 0; i < 4; ++i) { v8f z = {}; acc[i] = z; }
        gemm_chain<WT, true>(acc, bufA, 16, chart_w, 0, 16, col, lane);

        float cb[CDIM];                       // hoist chart_b column loads
#pragma unroll
        for (int c = 0; c < CDIM; ++c) cb[c] = chart_b[c * HDIM + col];
#pragma unroll
        for (int mt = 0; mt < 4; ++mt) {
#pragma unroll
            for (int j = 0; j < 8; ++j) {
                const int m = mt * 16 + 8 * half + j;
                float h = acc[mt][j];
#pragma unroll
                for (int c = 0; c < CDIM; ++c) h += r_f[m * CDIM + c] * cb[c];
                h_bf[swz16(m, col, 16)]  = f2bf(h);
                g1_bf[swz16(m, col, 16)] = f2bf(gelu_exact(h));
            }
        }
    }
    __syncthreads();

    // ---- Phase 2: h1 = gelu(g1 @ mlp1_w.T + b1)  (h1 overwrites bufA) ----
#pragma unroll 1
    for (int tt = 0; tt < NTPW; ++tt) {
        const int col = ((wv * NTPW + tt) << 4) + nn;
        v8f acc[4];
#pragma unroll
        for (int i = 0; i < 4; ++i) { v8f z = {}; acc[i] = z; }
        gemm_chain<WT, false>(acc, g1_bf, 16, mlp1_w, HDIM, 16, col, lane);

        const float bias = mlp1_b[col];
#pragma unroll
        for (int mt = 0; mt < 4; ++mt) {
#pragma unroll
            for (int j = 0; j < 8; ++j) {
                const int m = mt * 16 + 8 * half + j;
                bufA[swz16(m, col, 16)] = f2bf(gelu_exact(acc[mt][j] + bias));
            }
        }
    }
    __syncthreads();

    // ---- Phase 3: x = h1@mlp2^T + h@skip^T + ts*(zt@tex^T) + biases ----
#pragma unroll 1
    for (int tt = 0; tt < NTPW; ++tt) {
        const int col = ((wv * NTPW + tt) << 4) + nn;
        v8f acc[4];
#pragma unroll
        for (int i = 0; i < 4; ++i) { v8f z = {}; acc[i] = z; }
        // tex residual first (K = 64), then pre-scale so one accumulator serves all
        gemm_chain<WT, false>(acc, zt_bf, 2, tex_w, LDIM, 2, col, lane);
#pragma unroll
        for (int mt = 0; mt < 4; ++mt)
#pragma unroll
            for (int j = 0; j < 8; ++j) acc[mt][j] *= tscale;
        gemm_chain<WT, false>(acc, bufA, 16, mlp2_w, HDIM, 16, col, lane);  // mlp2 (A=h1)
        gemm_chain<WT, false>(acc, h_bf, 16, skip_w, HDIM, 16, col, lane);  // skip (A=h)

        const float bconst = mlp2_b[col] + skip_b[col] + tscale * tex_b[col];
#pragma unroll
        for (int mt = 0; mt < 4; ++mt) {
#pragma unroll
            for (int j = 0; j < 8; ++j) {
                const int m = mt * 16 + 8 * half + j;
                out_x[(size_t)(row0 + m) * ODIM + col] = acc[mt][j] + bconst;
            }
        }
    }
}

extern "C" void kernel_launch(void* const* d_in, const int* in_sizes, int n_in,
                              void* d_out, int out_size, void* d_ws, size_t ws_size,
                              hipStream_t stream) {
    const float* z_geo    = (const float*)d_in[0];
    const float* z_tex    = (const float*)d_in[1];
    const float* chart_w  = (const float*)d_in[2];
    const float* chart_b  = (const float*)d_in[3];
    const float* router_w = (const float*)d_in[4];
    const float* router_b = (const float*)d_in[5];
    const float* mlp1_w   = (const float*)d_in[6];
    const float* mlp1_b   = (const float*)d_in[7];
    const float* mlp2_w   = (const float*)d_in[8];
    const float* mlp2_b   = (const float*)d_in[9];
    const float* skip_w   = (const float*)d_in[10];
    const float* skip_b   = (const float*)d_in[11];
    const float* tex_w    = (const float*)d_in[12];
    const float* tex_b    = (const float*)d_in[13];
    const float* tex_sc   = (const float*)d_in[14];

    const int Btot = in_sizes[0] / LDIM;      // 32768
    float* out_x = (float*)d_out;
    float* out_r = out_x + (size_t)Btot * ODIM;

    const int    nblocks    = Btot / ROWS;    // 512 blocks of 64 rows
    const size_t smem_bytes = 225280;         // 220 KB dynamic LDS

    const int n_chart = CDIM * HDIM * LDIM;   // 262144
    const int n_mlp   = HDIM * HDIM;          // 262144
    const int n_tex   = ODIM * LDIM;          // 32768
    const size_t n_total = (size_t)n_chart + 3 * (size_t)n_mlp + n_tex;  // 1081344

    if (ws_size >= n_total * sizeof(unsigned short)) {
        // Pre-convert all weights to packed bf16 in workspace (RNE, same numerics)
        unsigned short* wb = (unsigned short*)d_ws;
        unsigned short* w_chart = wb;
        unsigned short* w_mlp1  = wb + n_chart;
        unsigned short* w_mlp2  = wb + n_chart + n_mlp;
        unsigned short* w_skip  = wb + n_chart + 2 * n_mlp;
        unsigned short* w_tex   = wb + n_chart + 3 * n_mlp;
        hipLaunchKernelGGL(convert_w_kernel, dim3((n_chart + 255) / 256), dim3(256), 0, stream, chart_w, w_chart, n_chart);
        hipLaunchKernelGGL(convert_w_kernel, dim3((n_mlp + 255) / 256),   dim3(256), 0, stream, mlp1_w,  w_mlp1,  n_mlp);
        hipLaunchKernelGGL(convert_w_kernel, dim3((n_mlp + 255) / 256),   dim3(256), 0, stream, mlp2_w,  w_mlp2,  n_mlp);
        hipLaunchKernelGGL(convert_w_kernel, dim3((n_mlp + 255) / 256),   dim3(256), 0, stream, skip_w,  w_skip,  n_mlp);
        hipLaunchKernelGGL(convert_w_kernel, dim3((n_tex + 255) / 256),   dim3(256), 0, stream, tex_w,   w_tex,   n_tex);

        hipFuncSetAttribute((const void*)&topo_decoder_kernel<unsigned short>,
                            hipFuncAttributeMaxDynamicSharedMemorySize, (int)smem_bytes);
        hipLaunchKernelGGL((topo_decoder_kernel<unsigned short>), dim3(nblocks), dim3(BLOCK_THREADS),
                           smem_bytes, stream,
                           z_geo, z_tex, w_chart, chart_b, router_w, router_b,
                           w_mlp1, mlp1_b, w_mlp2, mlp2_b, w_skip, skip_b,
                           w_tex, tex_b, tex_sc, out_x, out_r);
    } else {
        hipFuncSetAttribute((const void*)&topo_decoder_kernel<float>,
                            hipFuncAttributeMaxDynamicSharedMemorySize, (int)smem_bytes);
        hipLaunchKernelGGL((topo_decoder_kernel<float>), dim3(nblocks), dim3(BLOCK_THREADS),
                           smem_bytes, stream,
                           z_geo, z_tex, chart_w, chart_b, router_w, router_b,
                           mlp1_w, mlp1_b, mlp2_w, mlp2_b, skip_w, skip_b,
                           tex_w, tex_b, tex_sc, out_x, out_r);
    }
}
